// Transformer_10144712753843
// MI455X (gfx1250) — matrix operand order repaired and checked by reading, not compiled, not run on hardware
//
#include <hip/hip_runtime.h>

typedef __bf16 bf16_t;
typedef __attribute__((ext_vector_type(16))) __bf16 v16bf;
typedef __attribute__((ext_vector_type(8)))  __bf16 v8bf;
typedef __attribute__((ext_vector_type(4)))  __bf16 v4bf;
typedef __attribute__((ext_vector_type(8)))  float   v8f;

#define BATCH    4
#define T_SEQ    2048
#define D_MODEL  1024
#define N_HEADS  16
#define HEAD_DIM 64
#define VOCAB    32000
#define ROWS     (BATCH * T_SEQ)   // 8192

// ---------------------------------------------------------------- WMMA helpers

static __device__ __forceinline__ v8f wmma_bf16(v16bf a, v16bf b, v8f c) {
  // (neg_a, A, neg_b, B, c_mod, C, reuse_a, reuse_b)
  return __builtin_amdgcn_wmma_f32_16x16x32_bf16(false, a, false, b, (short)0, c,
                                                 false, false);
}

// A operand fragment (16x32 tile). `row` points at this lane's matrix row
// (m = lane&15) at the k-base of the 32-wide slice.
// element e -> k = (e>>3)*16 + ((lane>>4)*8) + (e&7)
static __device__ __forceinline__ v16bf load_a_frag(const bf16_t* row, int lane) {
  const int kb = ((lane >> 4) & 1) * 8;
  v8bf lo = *(const v8bf*)(row + kb);
  v8bf hi = *(const v8bf*)(row + kb + 16);
  v16bf r;
#pragma unroll
  for (int i = 0; i < 8; ++i) { r[i] = lo[i]; r[i + 8] = hi[i]; }
  return r;
}

// B operand fragment (32x16 tile). `row` points at this lane's column data
// (n = lane&15) stored k-major. element e -> k = (e>>3)*16 + ((lane>>4)*8) + (e&7)
static __device__ __forceinline__ v16bf load_b_frag(const bf16_t* row, int lane) {
  const int kb = ((lane >> 4) & 1) * 8;
  v8bf lo = *(const v8bf*)(row + kb);
  v8bf hi = *(const v8bf*)(row + kb + 16);
  v16bf r;
#pragma unroll
  for (int i = 0; i < 8; ++i) { r[i] = lo[i]; r[i + 8] = hi[i]; }
  return r;
}

// ------------------------------------------------- embedding + LayerNorm (f32)

__global__ __launch_bounds__(256) void embed_ln_kernel(
    const int* __restrict__ tokens, const float* __restrict__ tok_embed,
    const float* __restrict__ pos_embed, const float* __restrict__ ln_g,
    const float* __restrict__ ln_b, float* __restrict__ x_res,
    bf16_t* __restrict__ hout) {
  const int row = blockIdx.x;            // 0..8191
  const int t   = row & (T_SEQ - 1);
  const int tok = tokens[row];
  const int i   = threadIdx.x;           // 256 threads * float4 = 1024

  const float4 a = ((const float4*)(tok_embed + (size_t)tok * D_MODEL))[i];
  const float4 p = ((const float4*)(pos_embed + (size_t)t * D_MODEL))[i];
  float4 x;
  x.x = a.x + p.x; x.y = a.y + p.y; x.z = a.z + p.z; x.w = a.w + p.w;

  float s  = x.x + x.y + x.z + x.w;
  float ss = x.x * x.x + x.y * x.y + x.z * x.z + x.w * x.w;
#pragma unroll
  for (int off = 1; off < 32; off <<= 1) {
    s  += __shfl_xor(s, off, 32);
    ss += __shfl_xor(ss, off, 32);
  }
  __shared__ float sred[8], ssred[8];
  const int wid = i >> 5, lane = i & 31;
  if (lane == 0) { sred[wid] = s; ssred[wid] = ss; }
  __syncthreads();
  if (wid == 0) {
    float s2  = (lane < 8) ? sred[lane]  : 0.f;
    float ss2 = (lane < 8) ? ssred[lane] : 0.f;
#pragma unroll
    for (int off = 1; off < 8; off <<= 1) {
      s2  += __shfl_xor(s2, off, 32);
      ss2 += __shfl_xor(ss2, off, 32);
    }
    if (lane == 0) { sred[0] = s2; ssred[0] = ss2; }
  }
  __syncthreads();
  const float mu  = sred[0] * (1.0f / (float)D_MODEL);
  const float var = ssred[0] * (1.0f / (float)D_MODEL) - mu * mu;
  const float inv = rsqrtf(var + 1e-5f);

  ((float4*)(x_res + (size_t)row * D_MODEL))[i] = x;

  const float4 g = ((const float4*)ln_g)[i];
  const float4 b = ((const float4*)ln_b)[i];
  v4bf hv;
  hv[0] = (bf16_t)((x.x - mu) * inv * g.x + b.x);
  hv[1] = (bf16_t)((x.y - mu) * inv * g.y + b.y);
  hv[2] = (bf16_t)((x.z - mu) * inv * g.z + b.z);
  hv[3] = (bf16_t)((x.w - mu) * inv * g.w + b.w);
  *(v4bf*)&hout[(size_t)row * D_MODEL + i * 4] = hv;
}

// --------------------------------------- f32 -> bf16 transpose (build Bt = W^T)

__global__ __launch_bounds__(256) void transpose_kernel(
    const float* __restrict__ src, bf16_t* __restrict__ dst, int R, int C) {
  __shared__ float tile[32][33];
  const size_t base = (size_t)blockIdx.z * (size_t)R * (size_t)C;
  const int c0 = blockIdx.x * 32, r0 = blockIdx.y * 32;
  const int tx = threadIdx.x, ty = threadIdx.y;  // (32, 8)
#pragma unroll
  for (int j = 0; j < 32; j += 8)
    tile[ty + j][tx] = src[base + (size_t)(r0 + ty + j) * C + (c0 + tx)];
  __syncthreads();
#pragma unroll
  for (int j = 0; j < 32; j += 8)
    dst[base + (size_t)(c0 + ty + j) * R + (r0 + tx)] = (bf16_t)tile[tx][ty + j];
}

// ----------------------------------------------------- WMMA bf16 GEMM (3 modes)
// C[M,N] = A[M,K] * Bt[N,K]^T  ; M,N multiples of 128, K multiple of 32.
// MODE 0: QKV scatter  MODE 1: +residual -> bf16  MODE 2: +bias -> f32 (logits)

template <int MODE>
__global__ __launch_bounds__(256) void gemm_kernel(
    const bf16_t* __restrict__ A, const bf16_t* __restrict__ Bt,
    int M, int N, int K,
    float* __restrict__ outF, const float* __restrict__ bias,
    const float* __restrict__ resid,
    bf16_t* __restrict__ qO, bf16_t* __restrict__ kO, bf16_t* __restrict__ vO,
    bf16_t* __restrict__ outBf) {
  __shared__ bf16_t As[128][40];   // 80B row stride: conflict-free b128 reads
  __shared__ bf16_t Bs[128][40];

  const int tid  = threadIdx.x;
  const int lane = tid & 31;
  const int wid  = tid >> 5;
  const int wm   = wid & 1;        // 2 waves in M
  const int wn   = wid >> 1;       // 4 waves in N
  const int m0   = blockIdx.y * 128;
  const int n0   = blockIdx.x * 128;

  v8f acc[4][2] = {};

  for (int k0 = 0; k0 < K; k0 += 32) {
#pragma unroll
    for (int it = 0; it < 2; ++it) {
      const int idx = tid + it * 256;
      const int r = idx >> 2, seg = (idx & 3) * 8;
      *(v8bf*)&As[r][seg] = *(const v8bf*)&A[(size_t)(m0 + r) * K + k0 + seg];
      *(v8bf*)&Bs[r][seg] = *(const v8bf*)&Bt[(size_t)(n0 + r) * K + k0 + seg];
      if (k0 + 32 < K) {
        __builtin_prefetch(&A[(size_t)(m0 + r) * K + k0 + 32 + seg], 0, 0);
        __builtin_prefetch(&Bt[(size_t)(n0 + r) * K + k0 + 32 + seg], 0, 0);
      }
    }
    __syncthreads();

    v16bf af[4], bfr[2];
#pragma unroll
    for (int im = 0; im < 4; ++im)
      af[im] = load_a_frag(&As[wm * 64 + im * 16 + (lane & 15)][0], lane);
#pragma unroll
    for (int in = 0; in < 2; ++in)
      bfr[in] = load_b_frag(&Bs[wn * 32 + in * 16 + (lane & 15)][0], lane);
#pragma unroll
    for (int im = 0; im < 4; ++im)
#pragma unroll
      for (int in = 0; in < 2; ++in)
        acc[im][in] = wmma_bf16(af[im], bfr[in], acc[im][in]);
    __syncthreads();
  }

  // ---- epilogue (C/D layout: m = v + 8*(lane>>4), n = lane&15) ----
#pragma unroll
  for (int im = 0; im < 4; ++im)
#pragma unroll
    for (int in = 0; in < 2; ++in) {
      const int mB = m0 + wm * 64 + im * 16 + 8 * (lane >> 4);
      const int n  = n0 + wn * 32 + in * 16 + (lane & 15);
#pragma unroll
      for (int v = 0; v < 8; ++v) {
        const int m = mB + v;
        const float val = acc[im][in][v];
        if (MODE == 0) {  // QKV scatter: n = qkv*1024 + h*64 + hd
          const int sel = n >> 10;
          const int hh  = (n >> 6) & 15;
          const int hd  = n & 63;
          const size_t head = (size_t)((m >> 11) * N_HEADS + hh);
          const int tt = m & (T_SEQ - 1);
          if (sel == 0)
            qO[(head * T_SEQ + tt) * HEAD_DIM + hd] = (bf16_t)val;
          else if (sel == 1)
            kO[(head * T_SEQ + tt) * HEAD_DIM + hd] = (bf16_t)val;
          else  // V stored transposed per head: [head][hd][t]
            vO[(head * HEAD_DIM + hd) * T_SEQ + tt] = (bf16_t)val;
        } else if (MODE == 1) {  // residual add, bf16 out
          const size_t idx = (size_t)m * N + n;
          outBf[idx] = (bf16_t)(resid[idx] + val);
        } else {  // logits
          outF[(size_t)m * N + n] = val + bias[n];
        }
      }
    }
}

// -------------------------------------------------- flash attention (per b,h)
// grid: (T/128, B*H). Each wave owns 16 q rows; K/V^T tiles (64 wide) in LDS.

__global__ __launch_bounds__(256) void flash_attn_kernel(
    const bf16_t* __restrict__ qg, const bf16_t* __restrict__ kg,
    const bf16_t* __restrict__ vTg, bf16_t* __restrict__ attn) {
  __shared__ bf16_t Kt[64][72];        // [s][k]  == Bt for Q*K^T
  __shared__ bf16_t Vt[64][72];        // [hd][s] == Bt for P*V
  __shared__ bf16_t Pt[8][16][72];     // per-wave P (16 x 64), bf16

  const int bh    = blockIdx.y;                 // b*16 + h
  const int b     = bh >> 4;
  const int h     = bh & 15;
  const int qbase = blockIdx.x * 128;
  const int tid   = threadIdx.x;
  const int lane  = tid & 31;
  const int wid   = tid >> 5;

  // A fragments of Q (rows fixed for the whole kernel)
  const bf16_t* qrow =
      qg + ((size_t)bh * T_SEQ + qbase + wid * 16 + (lane & 15)) * HEAD_DIM;
  const v16bf aq0 = load_a_frag(qrow, lane);        // k 0..31
  const v16bf aq1 = load_a_frag(qrow + 32, lane);   // k 32..63

  v8f   o[4] = {};
  float mrow[8], lrow[8];
#pragma unroll
  for (int v = 0; v < 8; ++v) { mrow[v] = -1e30f; lrow[v] = 0.f; }

  const int qrow_base = qbase + wid * 16 + 8 * (lane >> 4);

  for (int s0 = 0; s0 < qbase + 128; s0 += 64) {
    // -------- stage K tile [64][64] and V^T tile [64][64] --------
#pragma unroll
    for (int it = 0; it < 2; ++it) {
      const int idx = tid + it * 256;
      const int r = idx >> 3, seg = (idx & 7) * 8;
      *(v8bf*)&Kt[r][seg] =
          *(const v8bf*)&kg[((size_t)bh * T_SEQ + s0 + r) * HEAD_DIM + seg];
      *(v8bf*)&Vt[r][seg] =
          *(const v8bf*)&vTg[((size_t)bh * HEAD_DIM + r) * T_SEQ + s0 + seg];
    }
    __syncthreads();

    // -------- scores = Q * K^T (16 x 64), scaled + causal mask --------
    v8f st[4];
#pragma unroll
    for (int jn = 0; jn < 4; ++jn) {
      const bf16_t* krow = &Kt[jn * 16 + (lane & 15)][0];
      v8f a = {};
      a = wmma_bf16(aq0, load_b_frag(krow, lane), a);
      a = wmma_bf16(aq1, load_b_frag(krow + 32, lane), a);
      st[jn] = a;
    }
#pragma unroll
    for (int jn = 0; jn < 4; ++jn) {
      const int sg = s0 + jn * 16 + (lane & 15);
#pragma unroll
      for (int v = 0; v < 8; ++v) {
        const float val = st[jn][v] * 0.125f;  // 1/sqrt(64)
        st[jn][v] = (sg > qrow_base + v) ? -3.0e38f : val;
      }
    }

    // -------- online softmax (rows live in 16-lane halves) --------
    float mnew[8], alpha[8], psum[8];
#pragma unroll
    for (int v = 0; v < 8; ++v) {
      float mx = fmaxf(fmaxf(st[0][v], st[1][v]), fmaxf(st[2][v], st[3][v]));
#pragma unroll
      for (int off = 1; off < 16; off <<= 1) mx = fmaxf(mx, __shfl_xor(mx, off, 32));
      mnew[v]  = fmaxf(mrow[v], mx);
      alpha[v] = __expf(mrow[v] - mnew[v]);
      psum[v]  = 0.f;
    }
#pragma unroll
    for (int jn = 0; jn < 4; ++jn)
#pragma unroll
      for (int v = 0; v < 8; ++v) {
        const float p = __expf(st[jn][v] - mnew[v]);
        st[jn][v] = p;
        psum[v] += p;
      }
#pragma unroll
    for (int v = 0; v < 8; ++v) {
#pragma unroll
      for (int off = 1; off < 16; off <<= 1) psum[v] += __shfl_xor(psum[v], off, 32);
      lrow[v] = lrow[v] * alpha[v] + psum[v];
      mrow[v] = mnew[v];
    }
#pragma unroll
    for (int jn = 0; jn < 4; ++jn)
#pragma unroll
      for (int v = 0; v < 8; ++v) o[jn][v] *= alpha[v];

    // -------- P (D-layout) -> LDS -> A-fragments --------
    bf16_t* pw = &Pt[wid][0][0];
#pragma unroll
    for (int jn = 0; jn < 4; ++jn)
#pragma unroll
      for (int v = 0; v < 8; ++v)
        pw[(v + 8 * (lane >> 4)) * 72 + jn * 16 + (lane & 15)] = (bf16_t)st[jn][v];

    const bf16_t* prow = &Pt[wid][lane & 15][0];
    const v16bf pf0 = load_a_frag(prow, lane);        // s 0..31
    const v16bf pf1 = load_a_frag(prow + 32, lane);   // s 32..63

    // -------- O += P * V --------
#pragma unroll
    for (int jn = 0; jn < 4; ++jn) {
      const bf16_t* vrow = &Vt[jn * 16 + (lane & 15)][0];
      o[jn] = wmma_bf16(pf0, load_b_frag(vrow, lane), o[jn]);
      o[jn] = wmma_bf16(pf1, load_b_frag(vrow + 32, lane), o[jn]);
    }
    __syncthreads();
  }

  // -------- normalize and store (concat heads, row-major [8192,1024]) --------
#pragma unroll
  for (int jn = 0; jn < 4; ++jn)
#pragma unroll
    for (int v = 0; v < 8; ++v) {
      const int m   = qrow_base + v;
      const int col = h * HEAD_DIM + jn * 16 + (lane & 15);
      attn[((size_t)b * T_SEQ + m) * D_MODEL + col] = (bf16_t)(o[jn][v] / lrow[v]);
    }
}

// ---------------------------------------------------------------------- launch

extern "C" void kernel_launch(void* const* d_in, const int* in_sizes, int n_in,
                              void* d_out, int out_size, void* d_ws, size_t ws_size,
                              hipStream_t stream) {
  (void)in_sizes; (void)n_in; (void)out_size; (void)ws_size;

  const int*   tokens  = (const int*)d_in[0];
  const float* tok_emb = (const float*)d_in[1];
  const float* pos_emb = (const float*)d_in[2];
  const float* ln_g    = (const float*)d_in[3];
  const float* ln_b    = (const float*)d_in[4];
  const float* Wq      = (const float*)d_in[5];
  const float* Wk      = (const float*)d_in[6];
  const float* Wv      = (const float*)d_in[7];
  const float* Wproj   = (const float*)d_in[8];
  const float* fc_w    = (const float*)d_in[9];
  const float* fc_b    = (const float*)d_in[10];
  float* out = (float*)d_out;

  char* ws = (char*)d_ws;
  size_t off = 0;
  auto take = [&](size_t bytes) -> char* {
    char* p = ws + off;
    off = (off + bytes + 255) & ~(size_t)255;
    return p;
  };
  float*  x_res  = (float*) take((size_t)ROWS * D_MODEL * 4);   // residual f32
  bf16_t* hbuf   = (bf16_t*)take((size_t)ROWS * D_MODEL * 2);   // LN(x) bf16
  bf16_t* qbuf   = (bf16_t*)take((size_t)ROWS * D_MODEL * 2);   // [b,h,t,hd]
  bf16_t* kbuf   = (bf16_t*)take((size_t)ROWS * D_MODEL * 2);   // [b,h,s,hd]
  bf16_t* vTbuf  = (bf16_t*)take((size_t)ROWS * D_MODEL * 2);   // [b,h,hd,s]
  bf16_t* attnb  = (bf16_t*)take((size_t)ROWS * D_MODEL * 2);   // concat heads
  bf16_t* x2buf  = (bf16_t*)take((size_t)ROWS * D_MODEL * 2);   // x + proj, bf16
  bf16_t* wqkvT  = (bf16_t*)take((size_t)3 * D_MODEL * D_MODEL * 2); // [3072,1024]
  bf16_t* wprojT = (bf16_t*)take((size_t)D_MODEL * D_MODEL * 2);
  bf16_t* fcT    = (bf16_t*)take((size_t)VOCAB * D_MODEL * 2);  // [32000,1024]

  const dim3 tb(32, 8);
  // Wq/Wk/Wv: per-head [D,HD] -> [HD,D]; concatenated rows: qkv*1024 + h*64 + hd
  transpose_kernel<<<dim3(HEAD_DIM / 32, D_MODEL / 32, N_HEADS), tb, 0, stream>>>(
      Wq, wqkvT, D_MODEL, HEAD_DIM);
  transpose_kernel<<<dim3(HEAD_DIM / 32, D_MODEL / 32, N_HEADS), tb, 0, stream>>>(
      Wk, wqkvT + (size_t)D_MODEL * D_MODEL, D_MODEL, HEAD_DIM);
  transpose_kernel<<<dim3(HEAD_DIM / 32, D_MODEL / 32, N_HEADS), tb, 0, stream>>>(
      Wv, wqkvT + (size_t)2 * D_MODEL * D_MODEL, D_MODEL, HEAD_DIM);
  transpose_kernel<<<dim3(D_MODEL / 32, D_MODEL / 32, 1), tb, 0, stream>>>(
      Wproj, wprojT, D_MODEL, D_MODEL);
  transpose_kernel<<<dim3(VOCAB / 32, D_MODEL / 32, 1), tb, 0, stream>>>(
      fc_w, fcT, D_MODEL, VOCAB);

  embed_ln_kernel<<<ROWS, 256, 0, stream>>>(tokens, tok_emb, pos_emb, ln_g, ln_b,
                                            x_res, hbuf);

  // QKV: [8192,1024] x [1024,3072]
  gemm_kernel<0><<<dim3(3 * D_MODEL / 128, ROWS / 128), 256, 0, stream>>>(
      hbuf, wqkvT, ROWS, 3 * D_MODEL, D_MODEL,
      nullptr, nullptr, nullptr, qbuf, kbuf, vTbuf, nullptr);

  flash_attn_kernel<<<dim3(T_SEQ / 128, BATCH * N_HEADS), 256, 0, stream>>>(
      qbuf, kbuf, vTbuf, attnb);

  // proj + residual: [8192,1024] x [1024,1024], += x_res, -> bf16
  gemm_kernel<1><<<dim3(D_MODEL / 128, ROWS / 128), 256, 0, stream>>>(
      attnb, wprojT, ROWS, D_MODEL, D_MODEL,
      nullptr, nullptr, x_res, nullptr, nullptr, nullptr, x2buf);

  // logits: [8192,1024] x [1024,32000] + fc_b -> f32 out
  gemm_kernel<2><<<dim3(VOCAB / 128, ROWS / 128), 256, 0, stream>>>(
      x2buf, fcT, ROWS, VOCAB, D_MODEL,
      out, fc_b, nullptr, nullptr, nullptr, nullptr, nullptr);
}
